// LSTM_85521388798687
// MI455X (gfx1250) — compile-verified, matrix operand order checked by reading, not run on hardware
//
#include <hip/hip_runtime.h>
#include <math.h>

// ---------------------------------------------------------------------------
// LSTM on gfx1250: persistent WMMA kernel, bf16 matrix cores, fp32 accumulate.
// N=32, T=1024, D=512, H=512, 4H=2048.
// ---------------------------------------------------------------------------

typedef __attribute__((ext_vector_type(16))) __bf16 bf16x16;
typedef __attribute__((ext_vector_type(8)))  float  f32x8;

#define LSTM_N   32
#define LSTM_T   1024
#define LSTM_D   512
#define LSTM_H   512
#define LSTM_4H  2048
#define KCHUNKS  16          // 512 / 32
#define NBLOCKS  16          // persistent grid
#define WAVES_PB 4           // 128 threads / block

// ---------------------------------------------------------------------------
// Pack a (K=512 x Ncol=2048) fp32 weight matrix into bf16 WMMA B-fragment
// tile order. Tile (ntp, kc) covers cols [ntp*16, +16), K [kc*32, +32).
// Within a tile, lane l's 16 elements (32 bytes, contiguous) are
//   W[kc*32 + 16*(l>>4) + e][ntp*16 + (l&15)]   e = 0..15
// i.e. contiguous k order (k = 16h + e), the same order the A fragments use.
// ---------------------------------------------------------------------------
__global__ void lstm_pack_w(const float* __restrict__ W, __bf16* __restrict__ P) {
    const int wid  = (blockIdx.x * blockDim.x + threadIdx.x) >> 5;  // tile id
    const int lane = threadIdx.x & 31;
    if (wid >= (LSTM_4H / 16) * KCHUNKS) return;
    const int ntp = wid >> 4;     // 0..127  (column tile in 4H)
    const int kc  = wid & 15;     // 0..15   (k chunk)
    const int col  = ntp * 16 + (lane & 15);
    const int krow = kc * 32 + ((lane >> 4) << 4);
    __bf16* dst = P + (size_t)wid * 512 + (size_t)lane * 16;
    const float* src = W + (size_t)krow * LSTM_4H + col;
#pragma unroll
    for (int e = 0; e < 16; ++e)
        dst[e] = (__bf16)src[(size_t)e * LSTM_4H];
}

// h0 (fp32, row-major 32x512) -> bf16 state buffer; also reset the barrier.
__global__ void lstm_init(const float* __restrict__ h0,
                          __bf16* __restrict__ hbuf,
                          unsigned* __restrict__ bar) {
    const int i = blockIdx.x * blockDim.x + threadIdx.x;
    if (i < LSTM_N * LSTM_H) hbuf[i] = (__bf16)h0[i];
    if (i == 0) { bar[0] = 0u; bar[1] = 0u; }
}

// ---------------------------------------------------------------------------
// Device-wide sense barrier (16 blocks). Release/acquire fences on both sides.
// ---------------------------------------------------------------------------
__device__ __forceinline__ void grid_barrier(unsigned* cnt, unsigned* gen) {
    __threadfence();                 // release: my writes visible device-wide
    __syncthreads();
    if (threadIdx.x == 0) {
        unsigned g = __atomic_load_n(gen, __ATOMIC_RELAXED);
        unsigned old = __atomic_fetch_add(cnt, 1u, __ATOMIC_ACQ_REL);
        if (old == NBLOCKS - 1) {
            __atomic_store_n(cnt, 0u, __ATOMIC_RELAXED);
            __atomic_fetch_add(gen, 1u, __ATOMIC_RELEASE);
        } else {
            while (__atomic_load_n(gen, __ATOMIC_ACQUIRE) == g)
                __builtin_amdgcn_s_sleep(1);
        }
    }
    __syncthreads();
    __threadfence();                 // acquire: drop stale cached h
}

__device__ __forceinline__ float sigmoidf_fast(float x) {
    return 1.0f / (1.0f + __expf(-x));
}

// ---------------------------------------------------------------------------
// Persistent recurrent kernel. 64 waves; wave (mt, nt) computes the 16x16
// tile (rows mt*16.., H-cols nt*16..) of all 4 gates each timestep:
//   a_g = x_t @ Wx[:, g*H + cols] + h_t @ Wh[:, g*H + cols] + b
// via 16 k-chunks x (2 A x 4 B) = 128 v_wmma_f32_16x16x32_bf16 per step.
// Cell state c stays in registers for the whole scan.
// 16 blocks x 4 waves = 1 wave/SIMD on 16 WGPs -> force single-occupancy
// register allocation (kills the scratch spills seen at default budget).
// ---------------------------------------------------------------------------
__global__ void __launch_bounds__(128, 1)
__attribute__((amdgpu_waves_per_eu(1, 1)))
lstm_scan(const float* __restrict__ x,
          const float* __restrict__ bias,
          const __bf16* __restrict__ pWx,
          const __bf16* __restrict__ pWh,
          __bf16* __restrict__ hA,        // ping (holds h0 at t=0)
          __bf16* __restrict__ hB,        // pong
          float* __restrict__ out,        // (N, T, H) fp32
          unsigned* __restrict__ bar) {
    const int lane = threadIdx.x & 31;
    const int wid  = blockIdx.x * WAVES_PB + (threadIdx.x >> 5);   // 0..63
    const int mt   = wid & 1;      // row tile (batch rows 0-15 / 16-31)
    const int nt   = wid >> 1;     // H column tile, 0..31

    const int mrow = mt * 16 + (lane & 15);      // my A-fragment row
    const int ksel = (lane >> 4) << 4;           // 0 or 16 within k-chunk
    const int coln = nt * 16 + (lane & 15);      // my C-fragment column in H

    // Per-wave packed-weight base: tile(g, kc) lives at
    //   base + g*(32*16*512) + kc*512   (elements), lane slice included.
    const size_t wbase = (size_t)nt * KCHUNKS * 512 + (size_t)lane * 16;
    const __bf16* __restrict__ wxb = pWx + wbase;
    const __bf16* __restrict__ whb = pWh + wbase;
#define GATE_STRIDE ((size_t)32 * KCHUNKS * 512)   // 262144 elements per gate

    // bias per gate (constant over t, broadcast over rows in C layout)
    float bval[4];
#pragma unroll
    for (int g = 0; g < 4; ++g) bval[g] = bias[g * LSTM_H + coln];

    f32x8 creg = {};                             // c0 = 0, stays in registers

    for (int t = 0; t < LSTM_T; ++t) {
        const __bf16* __restrict__ hin  = (t & 1) ? hB : hA;
        __bf16* __restrict__       hout = (t & 1) ? hA : hB;

        f32x8 acc0, acc1, acc2, acc3;
#pragma unroll
        for (int e = 0; e < 8; ++e) {
            acc0[e] = bval[0]; acc1[e] = bval[1];
            acc2[e] = bval[2]; acc3[e] = bval[3];
        }

        const float*  xrow = x   + ((size_t)mrow * LSTM_T + t) * LSTM_D;
        const __bf16* hrow = hin + (size_t)mrow * LSTM_H;

        // Speculative prefetch of next timestep's x slice (contiguous at +D).
        if (t + 1 < LSTM_T) {
            __builtin_prefetch(xrow + LSTM_D + ksel, 0, 0);
            __builtin_prefetch(xrow + LSTM_D + ksel + 256, 0, 0);
        }

#pragma unroll 4
        for (int kc = 0; kc < KCHUNKS; ++kc) {
            const int kb = kc * 32 + ksel;

            // A fragment from x_t: 16 contiguous fp32 -> bf16
            const float4* xp = (const float4*)(xrow + kb);
            float4 x0 = xp[0], x1 = xp[1], x2 = xp[2], x3 = xp[3];
            bf16x16 ax;
            ax[0]  = (__bf16)x0.x; ax[1]  = (__bf16)x0.y;
            ax[2]  = (__bf16)x0.z; ax[3]  = (__bf16)x0.w;
            ax[4]  = (__bf16)x1.x; ax[5]  = (__bf16)x1.y;
            ax[6]  = (__bf16)x1.z; ax[7]  = (__bf16)x1.w;
            ax[8]  = (__bf16)x2.x; ax[9]  = (__bf16)x2.y;
            ax[10] = (__bf16)x2.z; ax[11] = (__bf16)x2.w;
            ax[12] = (__bf16)x3.x; ax[13] = (__bf16)x3.y;
            ax[14] = (__bf16)x3.z; ax[15] = (__bf16)x3.w;

            // A fragment from h_t: 16 contiguous bf16 (32 bytes)
            bf16x16 ah = *(const bf16x16*)(hrow + kb);

            const size_t ko = (size_t)kc * 512;
#define GATE(GI, ACC)                                                         \
            {                                                                 \
                bf16x16 bx = *(const bf16x16*)(wxb + GI * GATE_STRIDE + ko);  \
                ACC = __builtin_amdgcn_wmma_f32_16x16x32_bf16(                \
                    false, ax, false, bx, (short)0, ACC, false, false);       \
                bf16x16 bh = *(const bf16x16*)(whb + GI * GATE_STRIDE + ko);  \
                ACC = __builtin_amdgcn_wmma_f32_16x16x32_bf16(                \
                    false, ah, false, bh, (short)0, ACC, false, false);       \
            }
            GATE(0, acc0)
            GATE(1, acc1)
            GATE(2, acc2)
            GATE(3, acc3)
#undef GATE
        }

        // gates + state update on the C layout:
        // VGPR e, lanes 0-15: row e;  lanes 16-31: row e+8. col = lane&15.
#pragma unroll
        for (int e = 0; e < 8; ++e) {
            float ig = sigmoidf_fast(acc0[e]);
            float fg = sigmoidf_fast(acc1[e]);
            float og = sigmoidf_fast(acc2[e]);
            float gg = tanhf(acc3[e]);
            float cn = fg * creg[e] + ig * gg;
            creg[e] = cn;
            float hn = og * tanhf(cn);

            const int mg = mt * 16 + e + ((lane >> 4) << 3);   // batch row
            out[((size_t)mg * LSTM_T + t) * LSTM_H + coln] = hn;
            hout[(size_t)mg * LSTM_H + coln] = (__bf16)hn;
        }

        grid_barrier(&bar[0], &bar[1]);
    }
}

// ---------------------------------------------------------------------------
// Host side
// ---------------------------------------------------------------------------
extern "C" void kernel_launch(void* const* d_in, const int* in_sizes, int n_in,
                              void* d_out, int out_size, void* d_ws, size_t ws_size,
                              hipStream_t stream) {
    const float* x   = (const float*)d_in[0];   // (32, 1024, 512)
    const float* h0  = (const float*)d_in[1];   // (32, 512)
    const float* Wx  = (const float*)d_in[2];   // (512, 2048)
    const float* Wh  = (const float*)d_in[3];   // (512, 2048)
    const float* b   = (const float*)d_in[4];   // (2048,)
    float* out = (float*)d_out;                 // (32, 1024, 512)

    char* ws = (char*)d_ws;
    __bf16* pWx = (__bf16*)(ws);                              // 2 MB
    __bf16* pWh = (__bf16*)(ws + (2u << 20));                 // 2 MB
    __bf16* hA  = (__bf16*)(ws + (4u << 20));                 // 32 KB
    __bf16* hB  = (__bf16*)(ws + (4u << 20) + (64u << 10));   // 32 KB
    unsigned* bar = (unsigned*)(ws + (4u << 20) + (128u << 10));

    // 2048 tiles per matrix, one wave per tile, 4 waves per block
    const int packBlocks = ((LSTM_4H / 16) * KCHUNKS) / WAVES_PB;   // 512
    lstm_pack_w<<<packBlocks, 128, 0, stream>>>(Wx, pWx);
    lstm_pack_w<<<packBlocks, 128, 0, stream>>>(Wh, pWh);
    lstm_init<<<(LSTM_N * LSTM_H + 255) / 256, 256, 0, stream>>>(h0, hA, bar);
    lstm_scan<<<NBLOCKS, 128, 0, stream>>>(x, b, pWx, pWh, hA, hB, out, bar);
}
